// CausalFlowModel_74414603371120
// MI455X (gfx1250) — matrix operand order repaired and checked by reading, not compiled, not run on hardware
//
#include <hip/hip_runtime.h>
#include <hip/hip_bf16.h>
#include <hip/hip_fp16.h>

typedef _Float16 half_t;
typedef _Float16 v16h __attribute__((ext_vector_type(16)));
typedef _Float16 v8h  __attribute__((ext_vector_type(8)));
typedef float    v8f  __attribute__((ext_vector_type(8)));

// ---------------- workspace layout (bytes) ----------------
#define WS_FRAG_OFF 0                         // 40 frags * 512 halfs * 2B = 40960
#define WS_A_OFF    65536                     // a_pre: 1024*64 f32 = 262144
#define WS_B_OFF    (WS_A_OFF + 262144)       // b_pre: 1024*64 f32 = 262144
#define WS_H16_OFF  (WS_B_OFF + 262144)       // H  f16: 1024*64*2 = 131072
#define WS_R16_OFF  (WS_H16_OFF + 131072)     // rnn out f16: 131072

#define NFRAG 40                              // 2 MLPs * 12 + final 16

// ---- fast transcendentals (branch-free; native V_TANH_F32 when available) ----
#if __has_builtin(__builtin_amdgcn_exp2f)
#define EXP2F_(x) __builtin_amdgcn_exp2f(x)
#else
#define EXP2F_(x) exp2f(x)
#endif
#if __has_builtin(__builtin_amdgcn_rcpf)
#define RCPF_(x) __builtin_amdgcn_rcpf(x)
#else
#define RCPF_(x) (1.0f / (x))
#endif

__device__ __forceinline__ float fast_sigmoid(float x) {
    return RCPF_(1.0f + EXP2F_(x * -1.4426950408889634f));
}
__device__ __forceinline__ float fast_tanh(float x) {
#if __has_builtin(__builtin_amdgcn_tanhf)
    return __builtin_amdgcn_tanhf(x);
#else
    float a = fabsf(x);
    float e = EXP2F_(a * -2.8853900817779268f);      // exp(-2a)
    float r = (1.0f - e) * RCPF_(1.0f + e);
    return copysignf(r, x);
#endif
}

__device__ __forceinline__ v8f wmma_f16(v16h a, v16h b, v8f c) {
    // D = A(16x32 f16) * B(32x16 f16) + C(16x16 f32)
    return __builtin_amdgcn_wmma_f32_16x16x32_f16(false, a, false, b, (short)0, c, false, false);
}

// A fragment from row-major f16 matrix (16 rows, stride in halfs).
// lanes 0-15: M=lane, halfs0-15 = K 0..15 ; lanes 16-31: K 16..31.
__device__ __forceinline__ v16h load_a_rowmajor(const half_t* z, int stride, int kt, int lane) {
    int m    = lane & 15;
    int base = m * stride + kt * 32 + ((lane >> 4) & 1) * 16;
    v16h a;
#pragma unroll
    for (int i = 0; i < 16; ++i) a[i] = z[base + i];
    return a;
}

// B fragment from fragment-staged buffer: lane-contiguous 16 halfs (2x ds_load_b128).
__device__ __forceinline__ v16h load_b_frag(const half_t* w, int frag, int lane) {
    const half_t* p = w + frag * 512 + lane * 16;
    v16h b;
#pragma unroll
    for (int i = 0; i < 16; ++i) b[i] = p[i];
    return b;
}

// ---------------- K1: stage weights as WMMA B fragments ----------------
// B element (k,n) = W[n][srcK]; lane layout: n = lane%16, k = kt*32 + (lane/16)*16 + h.
// Layer-1 K order is permuted to [x(0..63), t(64)] so activation staging is aligned.
__global__ void cf_prep_w(const float* __restrict__ xw1, const float* __restrict__ xw2,
                          const float* __restrict__ xw3, const float* __restrict__ uw1,
                          const float* __restrict__ uw2, const float* __restrict__ uw3,
                          const float* __restrict__ cw, half_t* __restrict__ wfrag) {
    int e = blockIdx.x * blockDim.x + threadIdx.x;
    if (e >= NFRAG * 512) return;
    int frag = e >> 9;
    int li   = e & 511;
    int lane = li >> 4, h = li & 15;
    const float* W; int outN, inK, nt, kt, isL1 = 0;
    if (frag < 24) {
        int mm = frag / 12, ff = frag % 12;
        const float* w1 = mm ? uw1 : xw1;
        const float* w2 = mm ? uw2 : xw2;
        const float* w3 = mm ? uw3 : xw3;
        if (ff < 6)      { nt = ff / 3; kt = ff % 3; W = w1; outN = 20; inK = 65; isL1 = 1; }
        else if (ff < 8) { nt = ff - 6; kt = 0;      W = w2; outN = 20; inK = 20;  }
        else             { nt = ff - 8; kt = 0;      W = w3; outN = 64; inK = 20;  }
    } else {
        int ff = frag - 24; nt = ff >> 2; kt = ff & 3; W = cw; outN = 64; inK = 128;
    }
    int N = nt * 16 + (lane & 15);
    int K = kt * 32 + (lane >> 4) * 16 + h;
    int srcK = K;
    if (isL1) srcK = (K < 64) ? K + 1 : (K == 64 ? 0 : -1);   // permuted feature order
    float v = (N < outN && srcK >= 0 && srcK < inK) ? W[N * inK + srcK] : 0.0f;
    wfrag[frag * 512 + li] = (half_t)v;
}

// ---------------- K2: precompute u-driven parts of the RNN ----------------
__global__ void cf_prep_ab(const float* __restrict__ u, const float* __restrict__ i2h_w,
                           const float* __restrict__ i2h_b, const float* __restrict__ h2o_w,
                           const float* __restrict__ h2o_b, float* __restrict__ a_pre,
                           float* __restrict__ b_pre) {
    int e = blockIdx.x * blockDim.x + threadIdx.x;
    if (e >= 2 * 1024 * 64) return;
    int which = e >> 16;
    int m = e & 65535;
    int k = m >> 6, i = m & 63;
    const float* w  = which ? h2o_w : i2h_w;
    const float* bb = which ? h2o_b : i2h_b;
    float acc = bb[i];
#pragma unroll
    for (int c = 0; c < 8; ++c) acc += u[k * 8 + c] * w[i * 72 + c];
    (which ? b_pre : a_pre)[k * 64 + i] = acc;
}

// ---------------- K3: the serial recurrence h <- tanh(a_k + Wh h) ----------------
__global__ __launch_bounds__(64) void cf_rnn(const float* __restrict__ i2h_w,
                                             const float* __restrict__ a_pre,
                                             half_t* __restrict__ h16) {
    __shared__ __align__(16) float hs[64];
    int i = threadIdx.x;
    float wh[64];
#pragma unroll
    for (int j = 0; j < 64; ++j) wh[j] = i2h_w[i * 72 + 8 + j];   // Wh row in registers
    hs[i] = 0.0f;
    __syncthreads();
    const float4* h4 = (const float4*)hs;
    for (int k = 0; k < 1024; ++k) {
        float acc = a_pre[k * 64 + i];
#pragma unroll
        for (int j = 0; j < 16; ++j) {                            // 16x ds_load_b128
            float4 hv = h4[j];
            acc += wh[4 * j + 0] * hv.x + wh[4 * j + 1] * hv.y +
                   wh[4 * j + 2] * hv.z + wh[4 * j + 3] * hv.w;
        }
        float hn = fast_tanh(acc);                                // native v_tanh_f32
        __syncthreads();
        hs[i] = hn;
        h16[k * 64 + i] = (half_t)hn;
        __syncthreads();
    }
}

// ---------------- K4: u_rnn_out = tanh(b + H @ Wo^T) via WMMA ----------------
__global__ __launch_bounds__(128) void cf_rnnout(const float* __restrict__ h2o_w,
                                                 const float* __restrict__ b_pre,
                                                 const half_t* __restrict__ h16,
                                                 half_t* __restrict__ r16) {
    int wid = threadIdx.x >> 5, lane = threadIdx.x & 31;
    int rt = blockIdx.x * 4 + wid;           // 64 row-tiles of 16
    if (rt >= 64) return;                    // wave-uniform
#pragma unroll
    for (int nt = 0; nt < 4; ++nt) {
        v8f c = {};
#pragma unroll
        for (int kt = 0; kt < 2; ++kt) {
            int m = rt * 16 + (lane & 15);
            int ab = m * 64 + kt * 32 + ((lane >> 4) & 1) * 16;
            v16h a;
#pragma unroll
            for (int q = 0; q < 16; ++q) a[q] = h16[ab + q];
            int n  = nt * 16 + (lane & 15);
            int kb = kt * 32 + ((lane >> 4) & 1) * 16;
            v16h b;
#pragma unroll
            for (int q = 0; q < 16; ++q) b[q] = (half_t)h2o_w[n * 72 + 8 + kb + q];
            c = wmma_f16(a, b, c);
        }
        int N = nt * 16 + (lane & 15);
#pragma unroll
        for (int r = 0; r < 8; ++r) {
            int M = rt * 16 + r + 8 * ((lane >> 4) & 1);
            r16[M * 64 + N] = (half_t)fast_tanh(c[r] + b_pre[M * 64 + N]);
        }
    }
}

// ---------------- K5: the big batched phase ----------------
__device__ __forceinline__ void mlp_block(half_t* z, half_t* f, int fcol,
                                          const half_t* wlds, int fragbase,
                                          const float* __restrict__ b1,
                                          const float* __restrict__ b2,
                                          const float* __restrict__ b3, int lane) {
    int N  = lane & 15;
    int hi = (lane >> 4) & 1;
    // L1: K=65(pad96, 3 ktiles), N=20(pad32, 2 ntiles)
    v8f c0 = {}, c1 = {};
#pragma unroll
    for (int kt = 0; kt < 3; ++kt) {
        v16h a = load_a_rowmajor(z, 96, kt, lane);
        c0 = wmma_f16(a, load_b_frag(wlds, fragbase + 0 * 3 + kt, lane), c0);
        c1 = wmma_f16(a, load_b_frag(wlds, fragbase + 1 * 3 + kt, lane), c1);
    }
    float bv0 = (N < 20) ? b1[N] : 0.0f;
    float bv1 = ((16 + N) < 20) ? b1[16 + N] : 0.0f;
#pragma unroll
    for (int r = 0; r < 8; ++r) {
        int M = r + 8 * hi;
        z[M * 96 + N]      = (half_t)fast_sigmoid(c0[r] + bv0);   // padded N cols are killed
        z[M * 96 + 16 + N] = (half_t)fast_sigmoid(c1[r] + bv1);   // by zero weight frags
    }
    asm volatile("s_wait_dscnt 0" ::: "memory");
    // L2: K=20(pad32), N=20(pad32)
    {
        v16h a = load_a_rowmajor(z, 96, 0, lane);
        c0 = (v8f){}; c1 = (v8f){};
        c0 = wmma_f16(a, load_b_frag(wlds, fragbase + 6, lane), c0);
        c1 = wmma_f16(a, load_b_frag(wlds, fragbase + 7, lane), c1);
    }
    bv0 = (N < 20) ? b2[N] : 0.0f;
    bv1 = ((16 + N) < 20) ? b2[16 + N] : 0.0f;
#pragma unroll
    for (int r = 0; r < 8; ++r) {
        int M = r + 8 * hi;
        z[M * 96 + N]      = (half_t)fast_sigmoid(c0[r] + bv0);
        z[M * 96 + 16 + N] = (half_t)fast_sigmoid(c1[r] + bv1);
    }
    asm volatile("s_wait_dscnt 0" ::: "memory");
    // L3: K=20(pad32), N=64 ; sigmoid belongs to "stacked" -> apply here
    {
        v16h a  = load_a_rowmajor(z, 96, 0, lane);
        v8f d0 = {}, d1 = {}, d2 = {}, d3 = {};
        d0 = wmma_f16(a, load_b_frag(wlds, fragbase + 8,  lane), d0);
        d1 = wmma_f16(a, load_b_frag(wlds, fragbase + 9,  lane), d1);
        d2 = wmma_f16(a, load_b_frag(wlds, fragbase + 10, lane), d2);
        d3 = wmma_f16(a, load_b_frag(wlds, fragbase + 11, lane), d3);
#pragma unroll
        for (int nt2 = 0; nt2 < 4; ++nt2) {
            v8f dd = nt2 == 0 ? d0 : nt2 == 1 ? d1 : nt2 == 2 ? d2 : d3;
            float bv = b3[nt2 * 16 + N];
#pragma unroll
            for (int r = 0; r < 8; ++r) {
                int M = r + 8 * hi;
                f[M * 128 + fcol + nt2 * 16 + N] = (half_t)fast_sigmoid(dd[r] + bv);
            }
        }
    }
}

__global__ __launch_bounds__(64) void cf_main(
    const float* __restrict__ t, const float* __restrict__ x,
    const float* __restrict__ xb1, const float* __restrict__ xb2, const float* __restrict__ xb3,
    const float* __restrict__ ub1, const float* __restrict__ ub2, const float* __restrict__ ub3,
    const float* __restrict__ cb, const half_t* __restrict__ wfrag_g,
    const half_t* __restrict__ r16, float* __restrict__ out, int ntiles) {
    __shared__ __align__(16) half_t wlds[NFRAG * 512];   // 40 KB weight fragments
    __shared__ __align__(16) half_t zbuf[2][16 * 96];    // per-wave activation staging
    __shared__ __align__(16) half_t fbuf[2][16 * 128];   // per-wave final-layer input
    int tid = threadIdx.x;
    {   // b128 weight preload
        uint4* dst = (uint4*)wlds;
        const uint4* src = (const uint4*)wfrag_g;
        for (int i = tid; i < NFRAG * 512 * 2 / 16; i += 64) dst[i] = src[i];
    }
    __syncthreads();

    int wid = tid >> 5, lane = tid & 31;
    half_t* z = zbuf[wid];
    half_t* f = fbuf[wid];
    int gwave  = blockIdx.x * 2 + wid;
    int nwaves = gridDim.x * 2;

    for (int tile = gwave; tile < ntiles; tile += nwaves) {
        int rbase = tile * 16;
        int r = lane & 15, p = (lane >> 4) & 1;
        // ---- stage z = [x(0..63), t(64), 0-pad] : 2 lanes/row, 32 cols each ----
        {
            const float4* xr4 = (const float4*)(x + (size_t)(rbase + r) * 64 + p * 32);
            v8h* zr = (v8h*)(z + r * 96 + p * 32);
#pragma unroll
            for (int q = 0; q < 4; ++q) {                 // 8x b128 load, 4x b128 ds store
                float4 f0 = xr4[2 * q], f1 = xr4[2 * q + 1];
                v8h hv;
                hv[0] = (half_t)f0.x; hv[1] = (half_t)f0.y; hv[2] = (half_t)f0.z; hv[3] = (half_t)f0.w;
                hv[4] = (half_t)f1.x; hv[5] = (half_t)f1.y; hv[6] = (half_t)f1.z; hv[7] = (half_t)f1.w;
                zr[q] = hv;
            }
            if (p) {                                      // t + zero pad, b128 stores
                float tv = t[rbase + r];
                v8h h0 = {}; h0[0] = (half_t)tv;
                v8h zz = {};
                v8h* ze = (v8h*)(z + r * 96 + 64);
                ze[0] = h0; ze[1] = zz; ze[2] = zz; ze[3] = zz;
            }
        }
        asm volatile("s_wait_dscnt 0" ::: "memory");
        mlp_block(z, f, 0, wlds, 0, xb1, xb2, xb3, lane);          // state branch
        // ---- stage z = [enc(0..63), t(64), 0-pad] via gathered f16 rows ----
        {
            float tv = t[rbase + r];
            int idx = (int)(tv * 1024.0f);
            idx = idx < 0 ? 0 : (idx > 1023 ? 1023 : idx);
            const v8h* er = (const v8h*)(r16 + (size_t)idx * 64) + p * 4;
            v8h* zr = (v8h*)(z + r * 96 + p * 32);
#pragma unroll
            for (int q = 0; q < 4; ++q) zr[q] = er[q];    // 4x b128 load -> b128 ds store
            if (p) {
                v8h h0 = {}; h0[0] = (half_t)tv;
                v8h zz = {};
                v8h* ze = (v8h*)(z + r * 96 + 64);
                ze[0] = h0; ze[1] = zz; ze[2] = zz; ze[3] = zz;
            }
        }
        asm volatile("s_wait_dscnt 0" ::: "memory");
        mlp_block(z, f, 64, wlds, 12, ub1, ub2, ub3, lane);        // control branch
        asm volatile("s_wait_dscnt 0" ::: "memory");
        // ---- final: out = sigmoid(stacked) @ cw^T + cb ; K=128, N=64 ----
#pragma unroll
        for (int nt = 0; nt < 4; ++nt) {
            v8f c = {};
#pragma unroll
            for (int kt = 0; kt < 4; ++kt) {
                v16h a = load_a_rowmajor(f, 128, kt, lane);
                c = wmma_f16(a, load_b_frag(wlds, 24 + nt * 4 + kt, lane), c);
            }
            int N = nt * 16 + (lane & 15);
            float bv = cb[N];
#pragma unroll
            for (int rr = 0; rr < 8; ++rr) {
                int M = rbase + rr + 8 * p;
                out[(size_t)M * 64 + N] = c[rr] + bv;              // lanes 0-15: coalesced 64B
            }
        }
    }
}

extern "C" void kernel_launch(void* const* d_in, const int* in_sizes, int n_in,
                              void* d_out, int out_size, void* d_ws, size_t ws_size,
                              hipStream_t stream) {
    const float* t     = (const float*)d_in[0];
    const float* x     = (const float*)d_in[1];
    const float* u     = (const float*)d_in[2];
    const float* i2h_w = (const float*)d_in[3];
    const float* i2h_b = (const float*)d_in[4];
    const float* h2o_w = (const float*)d_in[5];
    const float* h2o_b = (const float*)d_in[6];
    const float* xw1   = (const float*)d_in[7];
    const float* xb1   = (const float*)d_in[8];
    const float* xw2   = (const float*)d_in[9];
    const float* xb2   = (const float*)d_in[10];
    const float* xw3   = (const float*)d_in[11];
    const float* xb3   = (const float*)d_in[12];
    const float* uw1   = (const float*)d_in[13];
    const float* ub1   = (const float*)d_in[14];
    const float* uw2   = (const float*)d_in[15];
    const float* ub2   = (const float*)d_in[16];
    const float* uw3   = (const float*)d_in[17];
    const float* ub3   = (const float*)d_in[18];
    const float* cw    = (const float*)d_in[19];
    const float* cb    = (const float*)d_in[20];

    char*   ws    = (char*)d_ws;
    half_t* wfrag = (half_t*)(ws + WS_FRAG_OFF);
    float*  a_pre = (float*)(ws + WS_A_OFF);
    float*  b_pre = (float*)(ws + WS_B_OFF);
    half_t* h16   = (half_t*)(ws + WS_H16_OFF);
    half_t* r16   = (half_t*)(ws + WS_R16_OFF);

    int B      = in_sizes[0];     // 262144
    int ntiles = B / 16;

    cf_prep_w <<<(NFRAG * 512 + 255) / 256, 256, 0, stream>>>(xw1, xw2, xw3, uw1, uw2, uw3, cw, wfrag);
    cf_prep_ab<<<(2 * 1024 * 64 + 255) / 256, 256, 0, stream>>>(u, i2h_w, i2h_b, h2o_w, h2o_b, a_pre, b_pre);
    cf_rnn    <<<1, 64, 0, stream>>>(i2h_w, a_pre, h16);
    cf_rnnout <<<16, 128, 0, stream>>>(h2o_w, b_pre, h16, r16);
    cf_main   <<<2048, 64, 0, stream>>>(t, x, xb1, xb2, xb3, ub1, ub2, ub3, cb, wfrag, r16,
                                        (float*)d_out, ntiles);
}
